// MultiHeadAttention_6966436954437
// MI455X (gfx1250) — matrix operand order repaired and checked by reading, not compiled, not run on hardware
//
#include <hip/hip_runtime.h>
#include <hip/hip_bf16.h>
#include <stdint.h>

typedef __attribute__((ext_vector_type(16))) _Float16     v16h;
typedef __attribute__((ext_vector_type(8)))  float        v8f;
typedef __attribute__((ext_vector_type(4)))  unsigned int v4u;
typedef __attribute__((ext_vector_type(8)))  unsigned int v8u;

#define B_    4
#define S_    2048
#define D_    2048
#define H_    16
#define DH_   128
#define NQKV  6144   /* 3*H*DH, per-head [q(128) k(128) v(128)] slices */
#define HD_   2048   /* H*DH */

// Assemble a 16-half WMMA fragment from two 16-byte chunks (LDS or global).
__device__ __forceinline__ v16h frag_ld(const _Float16* p0, const _Float16* p1) {
    union { v16h h; v4u u[2]; } r;
    r.u[0] = *(const v4u*)p0;
    r.u[1] = *(const v4u*)p1;
    return r.h;
}

__device__ __forceinline__ uint32_t lds_addr32(const void* p) {
    return (uint32_t)(uintptr_t)p;   // flat->LDS uses addr[31:0]
}

// gfx1250 async global->LDS copy (16B), tracked by ASYNCcnt.
__device__ __forceinline__ void async_ld_b128(uint32_t lds_addr, const void* gptr) {
    asm volatile("global_load_async_to_lds_b128 %0, %1, off"
                 :: "v"(lds_addr), "v"(gptr) : "memory");
}
__device__ __forceinline__ void wait_async0() {
    asm volatile("s_wait_asynccnt 0" ::: "memory");
}

// gfx1250 Tensor Data Mover: DMA a 2D tile (16-bit elements, row-major) from
// global memory into LDS. Descriptor per CDNA5 ISA ch.8 (D# groups 0/1, 2D
// tensor -> groups 2/3 NULL). All inputs are wave-uniform -> SGPR operands.
__device__ __forceinline__ void tdm_load_2d_f16(uint32_t lds_addr, const void* gaddr,
                                                uint32_t tensor_w, uint32_t tensor_h,
                                                uint32_t tile_w,   uint32_t tile_h,
                                                uint32_t stride_elems)
{
    uint64_t ga = (uint64_t)(uintptr_t)gaddr;
    v4u g0;
    g0[0] = 1u;                                            // count=1, user mode
    g0[1] = lds_addr;                                      // lds_addr [63:32]
    g0[2] = (uint32_t)ga;                                  // global_addr lo
    g0[3] = (uint32_t)((ga >> 32) & 0x01FFFFFFu)           // global_addr[56:32]
            | (2u << 30);                                  // type=2 ("image")
    v8u g1;
    g1[0] = 1u << 16;                                      // data_size=1 (2B), wg_mask=0
    g1[1] = (tensor_w & 0xFFFFu) << 16;                    // tensor_dim0[15:0] @bit48
    g1[2] = (tensor_w >> 16) | ((tensor_h & 0xFFFFu) << 16); // dim0[31:16], dim1[15:0]
    g1[3] = (tensor_h >> 16) | (tile_w << 16);             // dim1[31:16], tile_dim0
    g1[4] = tile_h;                                        // tile_dim1 (tile_dim2=0)
    g1[5] = stride_elems;                                  // tensor_dim0_stride lo32
    g1[6] = 0u;                                            // stride hi16, dim1_stride lo
    g1[7] = 0u;
    asm volatile("tensor_load_to_lds %0, %1" :: "s"(g0), "s"(g1) : "memory");
}

// ---------------------------------------------------------------------------
// fp32 -> fp16 elementwise conversion
// ---------------------------------------------------------------------------
__global__ void cvt_f32_f16(const float* __restrict__ in, _Float16* __restrict__ out, int n) {
    int i = blockIdx.x * blockDim.x + threadIdx.x;
    int stride = gridDim.x * blockDim.x;
    for (; i < n; i += stride) out[i] = (_Float16)in[i];
}

// ---------------------------------------------------------------------------
// C[M,N] = A[M,K] * B[N,K]^T + bias[N]   (both operands row-major, K contig)
// Block: 64(M) x 128(N) tile, 128 threads = 4 waves, each wave 32x64
// (2x4 WMMA tiles) -> 8 wmma per K-step per wave.
// Tiles are fetched by the Tensor Data Mover (one descriptor per tile,
// issued by wave 0; TDM ignores EXEC so it must be wave-guarded).
// ---------------------------------------------------------------------------
__global__ __launch_bounds__(128)
void gemm_nt_wmma(const _Float16* __restrict__ A,
                  const _Float16* __restrict__ Bm,
                  const float*    __restrict__ bias,
                  _Float16* __restrict__ outH,   // fp16 out (or null)
                  float*    __restrict__ outF,   // fp32 out (or null)
                  int K, int ldOut, int Arows, int Brows)
{
    __shared__ __align__(16) _Float16 lA[64 * 32];    // 4 KB
    __shared__ __align__(16) _Float16 lB[128 * 32];   // 8 KB

    const int t    = threadIdx.x;
    const int lane = t & 31;
    const int wid  = t >> 5;
    const int lrow = lane & 15;
    const int lhi  = lane >> 4;

    const int mbase = blockIdx.y * 64;
    const int nbase = blockIdx.x * 128;
    const int waveM = (wid & 1) * 32;
    const int waveN = (wid >> 1) * 64;

    v8f acc[2][4] = {};

    for (int k0 = 0; k0 < K; k0 += 32) {
        __syncthreads();                     // previous readers done
        if (wid == 0) {
            // A tile: 64 rows x 32 halves, row stride K.
            tdm_load_2d_f16(lds_addr32(lA), A + (size_t)mbase * K + k0,
                            (uint32_t)(K - k0), (uint32_t)(Arows - mbase),
                            32u, 64u, (uint32_t)K);
            // B tile: 128 rows x 32 halves, row stride K.
            tdm_load_2d_f16(lds_addr32(lB), Bm + (size_t)nbase * K + k0,
                            (uint32_t)(K - k0), (uint32_t)(Brows - nbase),
                            32u, 128u, (uint32_t)K);
            __builtin_amdgcn_s_wait_tensorcnt(0);
        }
        if (k0 + 32 < K) {  // hint next K tile into cache (global_prefetch_b8)
            __builtin_prefetch(A + (size_t)(mbase + (t >> 1)) * K + k0 + 32, 0, 1);
            __builtin_prefetch(Bm + (size_t)(nbase + t) * K + k0 + 32, 0, 1);
        }
        __syncthreads();                     // TDM data visible to all waves

        // A fragments: lane = row (M), K pairs {0-7,16-23} / {8-15,24-31}
        v16h af[2], bf[4];
        const int akoff = lhi * 8;
        for (int i = 0; i < 2; ++i) {
            const _Float16* p = lA + (waveM + i * 16 + lrow) * 32 + akoff;
            af[i] = frag_ld(p, p + 16);
        }
        // B fragments: lane = col (N), K pairs {0-7,16-23} / {8-15,24-31}
        const int bkoff = lhi * 8;
        for (int j = 0; j < 4; ++j) {
            const _Float16* p = lB + (waveN + j * 16 + lrow) * 32 + bkoff;
            bf[j] = frag_ld(p, p + 16);
        }
        for (int i = 0; i < 2; ++i)
            for (int j = 0; j < 4; ++j)
                acc[i][j] = __builtin_amdgcn_wmma_f32_16x16x32_f16(
                    false, af[i], false, bf[j], (short)0, acc[i][j], false, false);
    }

    // Epilogue: D layout is VGPR r -> row r (lanes 0-15) / r+8 (lanes 16-31),
    // lane&15 -> col.
    for (int j = 0; j < 4; ++j) {
        int n = nbase + waveN + j * 16 + lrow;
        float bj = bias ? bias[n] : 0.0f;
        for (int i = 0; i < 2; ++i) {
            for (int r = 0; r < 8; ++r) {
                int m = mbase + waveM + i * 16 + r + lhi * 8;
                float v = acc[i][j][r] + bj;
                if (outH) outH[(size_t)m * ldOut + n] = (_Float16)v;
                else      outF[(size_t)m * ldOut + n] = v;
            }
        }
    }
}

// ---------------------------------------------------------------------------
// Causal flash attention. Block = (64 q-rows, head, batch), 4 waves x 16 rows.
// qkv: [B*S, 6144] fp16, per-head slice [q|k|v] of 128 each.
// ctx: [B*S, 2048] fp16.
// ---------------------------------------------------------------------------
__global__ __launch_bounds__(128)
void flash_attn_wmma(const _Float16* __restrict__ qkv,
                     _Float16* __restrict__ ctx)
{
    __shared__ __align__(16) _Float16 lK[32 * 128];      // K rows, d contiguous
    __shared__ __align__(16) _Float16 lVt[128 * 32];     // V transposed: [d][k]
    __shared__ __align__(16) _Float16 lP[4][16 * 32];    // per-wave P scratch

    const int t    = threadIdx.x;
    const int lane = t & 31;
    const int wid  = t >> 5;
    const int lrow = lane & 15;
    const int lhi  = lane >> 4;

    const int qb    = blockIdx.x * 64;         // q tile base in S
    const int h     = blockIdx.y;
    const int b     = blockIdx.z;
    const size_t rowbase = (size_t)b * S_;
    const int hq = h * 384, hk = h * 384 + 128, hv = h * 384 + 256;
    const int qrow0 = qb + wid * 16;           // this wave's first q row

    // Q fragments for the whole DH=128 (4 chunks of 32), loaded once.
    v16h aq[4];
    {
        const _Float16* qp = qkv + (rowbase + qrow0 + lrow) * NQKV + hq;
        for (int c = 0; c < 4; ++c) {
            const _Float16* p = qp + c * 32 + lhi * 8;
            aq[c] = frag_ld(p, p + 16);
        }
    }

    v8f cacc[8] = {};               // ctx accum: 8 d-tiles of 16
    float mrow[8], lsum[8];
    for (int r = 0; r < 8; ++r) { mrow[r] = -3.0e38f; lsum[r] = 0.0f; }

    const float scale = 0.08838834764831845f;  // 1/sqrt(128)
    const int nchunks = qb / 32 + 2;           // causal bound, uniform in block

    for (int kc = 0; kc < nchunks; ++kc) {
        const int kt = kc * 32;
        __syncthreads();
        // K tile 32x128 row-major via async global->LDS; V tile transposed
        // manually (element scatter into lVt).
        for (int i = 0; i < 4; ++i) {
            int lin = t + i * 128;            // 0..511
            int kk  = lin >> 4;               // 0..31 (K/V row)
            int seg = lin & 15;               // 8-half segment of d
            const size_t grow = (rowbase + kt + kk) * NQKV;
            async_ld_b128(lds_addr32(lK + kk * 128 + seg * 8),
                          qkv + grow + hk + seg * 8);
            union { v4u u; _Float16 hh[8]; } vv;
            vv.u = *(const v4u*)(qkv + grow + hv + seg * 8);
            for (int j = 0; j < 8; ++j)
                lVt[(seg * 8 + j) * 32 + kk] = vv.hh[j];
        }
        wait_async0();
        __syncthreads();

        // Scores: S[16q x 32k] = Q(16x128) * K(32x128)^T, two 16x16 D tiles.
        v8f s0 = {}, s1 = {};
        for (int c = 0; c < 4; ++c) {
            const _Float16* p0 = lK + lrow * 128 + c * 32 + lhi * 8;
            const _Float16* p1 = lK + (16 + lrow) * 128 + c * 32 + lhi * 8;
            v16h bk0 = frag_ld(p0, p0 + 16);
            v16h bk1 = frag_ld(p1, p1 + 16);
            s0 = __builtin_amdgcn_wmma_f32_16x16x32_f16(false, aq[c], false, bk0, (short)0, s0, false, false);
            s1 = __builtin_amdgcn_wmma_f32_16x16x32_f16(false, aq[c], false, bk1, (short)0, s1, false, false);
        }

        // Mask + online softmax. Row m = r + 8*lhi, col = lane&15 (+16 for s1).
        float p0v[8], p1v[8], corrv[8];
        for (int r = 0; r < 8; ++r) {
            int qg = qrow0 + r + 8 * lhi;
            float x0 = s0[r] * scale;
            float x1 = s1[r] * scale;
            if (kt + lrow > qg)      x0 = -__builtin_inff();
            if (kt + 16 + lrow > qg) x1 = -__builtin_inff();
            float mx = fmaxf(x0, x1);
            for (int off = 8; off; off >>= 1) mx = fmaxf(mx, __shfl_xor(mx, off, 32));
            float mnew = fmaxf(mrow[r], mx);      // floor -3e38, never -inf
            float corr = __expf(mrow[r] - mnew);
            float e0 = __expf(x0 - mnew);
            float e1 = __expf(x1 - mnew);
            float rs = e0 + e1;
            for (int off = 8; off; off >>= 1) rs += __shfl_xor(rs, off, 32);
            lsum[r] = lsum[r] * corr + rs;
            mrow[r] = mnew;
            corrv[r] = corr;
            p0v[r] = e0;
            p1v[r] = e1;
        }
        for (int tt = 0; tt < 8; ++tt)
            for (int r = 0; r < 8; ++r)
                cacc[tt][r] *= corrv[r];

        // P: D layout -> A layout via per-wave LDS bounce.
        _Float16* pw = lP[wid];
        for (int r = 0; r < 8; ++r) {
            pw[(r + 8 * lhi) * 32 + lrow]      = (_Float16)p0v[r];
            pw[(r + 8 * lhi) * 32 + 16 + lrow] = (_Float16)p1v[r];
        }
        const _Float16* pp = pw + lrow * 32 + lhi * 8;
        v16h ap = frag_ld(pp, pp + 16);

        // ctx += P(16x32) * V(32x128); B fragment reads lVt rows (d-major).
        for (int tt = 0; tt < 8; ++tt) {
            const _Float16* vp = lVt + (tt * 16 + lrow) * 32 + lhi * 8;
            v16h bv = frag_ld(vp, vp + 16);
            cacc[tt] = __builtin_amdgcn_wmma_f32_16x16x32_f16(
                false, ap, false, bv, (short)0, cacc[tt], false, false);
        }
    }

    // Normalize and write ctx [B*S, H*DH] fp16.
    for (int tt = 0; tt < 8; ++tt) {
        for (int r = 0; r < 8; ++r) {
            int m = qrow0 + r + 8 * lhi;
            float ov = cacc[tt][r] / lsum[r];
            ctx[(rowbase + m) * HD_ + h * DH_ + tt * 16 + lrow] = (_Float16)ov;
        }
    }
}

// ---------------------------------------------------------------------------
extern "C" void kernel_launch(void* const* d_in, const int* in_sizes, int n_in,
                              void* d_out, int out_size, void* d_ws, size_t ws_size,
                              hipStream_t stream) {
    const float* x      = (const float*)d_in[0];
    // d_in[1] = attn_mask (causal tril) — baked into the flash kernel
    const float* w_qkv  = (const float*)d_in[2];
    const float* b_qkv  = (const float*)d_in[3];
    const float* w_out  = (const float*)d_in[4];
    const float* b_out  = (const float*)d_in[5];
    float* out = (float*)d_out;

    char* ws = (char*)d_ws;
    _Float16* x_h   = (_Float16*)ws; ws += (size_t)B_ * S_ * D_ * 2;        // 32 MB
    _Float16* wq_h  = (_Float16*)ws; ws += (size_t)NQKV * D_ * 2;           // 24 MB
    _Float16* wo_h  = (_Float16*)ws; ws += (size_t)D_ * HD_ * 2;            // 8 MB
    _Float16* qkv_h = (_Float16*)ws; ws += (size_t)B_ * S_ * NQKV * 2;      // 96 MB
    _Float16* ctx_h = (_Float16*)ws;                                        // 32 MB

    const int nx = B_ * S_ * D_;      // 16.8M
    const int nq = NQKV * D_;         // 12.6M
    const int no = D_ * HD_;          // 4.2M
    cvt_f32_f16<<<4096, 256, 0, stream>>>(x,     x_h,  nx);
    cvt_f32_f16<<<4096, 256, 0, stream>>>(w_qkv, wq_h, nq);
    cvt_f32_f16<<<4096, 256, 0, stream>>>(w_out, wo_h, no);

    // qkv = x @ w_qkv^T + b_qkv : M=8192, N=6144, K=2048
    gemm_nt_wmma<<<dim3(NQKV / 128, (B_ * S_) / 64), 128, 0, stream>>>(
        x_h, wq_h, b_qkv, qkv_h, nullptr, D_, NQKV, B_ * S_, NQKV);

    // flash attention: grid = (S/64, H, B)
    flash_attn_wmma<<<dim3(S_ / 64, H_, B_), 128, 0, stream>>>(qkv_h, ctx_h);

    // out = ctx @ w_out^T + b_out : M=8192, N=2048, K=2048 (fp32 out)
    gemm_nt_wmma<<<dim3(D_ / 128, (B_ * S_) / 64), 128, 0, stream>>>(
        ctx_h, wo_h, b_out, nullptr, out, HD_, D_, B_ * S_, D_);
}
